// VQVAE_29927332118617
// MI455X (gfx1250) — matrix operand order repaired and checked by reading, not compiled, not run on hardware
//
#include <hip/hip_runtime.h>

typedef __attribute__((ext_vector_type(16))) _Float16 v16h;
typedef __attribute__((ext_vector_type(8)))  float    v8f;
typedef int a_v2i __attribute__((vector_size(8)));   // 8-byte payload type for async b64

#define BB 32   // batch

#if __has_builtin(__builtin_amdgcn_global_load_async_to_lds_b64) && \
    __has_builtin(__builtin_amdgcn_s_wait_asynccnt)
#define HAVE_ASYNC_LDS 1
#else
#define HAVE_ASYNC_LDS 0
#endif

// ---------------------------------------------------------------------------
// Generic implicit-GEMM conv / transposed-conv using WMMA f16 -> f32.
// GEMM: D[M=Cout][N=B*OH*OW] = Wpack[M][K] * im2col[K][N], K = Cin*KH*KW (padded to 32)
// Workgroup: 256 threads = 8 waves; tile 64(M) x 128(N).
// Wave (mw, half) -> A-subtile row mw, four 16-wide N subtiles => 4 WMMA / chunk.
// A tile staged with GLOBAL_LOAD_ASYNC_TO_LDS_B64 when available (ASYNCcnt),
// next-chunk weights prefetched with global_prefetch_b8.
// ---------------------------------------------------------------------------
__global__ __launch_bounds__(256)
void conv_wmma_k(const _Float16* __restrict__ in, const _Float16* __restrict__ wpack,
                 const float* __restrict__ bias, float* __restrict__ out,
                 int Cin, int H, int W, int Cout, int OH, int OW,
                 int KH, int KW, int stride, int pad, int Kreal, int Kpad, int deconv)
{
    const int S    = OH * OW;
    const int n0   = blockIdx.x * 128;
    const int m0   = blockIdx.y * 64;
    const int tid  = threadIdx.x;
    const int lane = tid & 31;
    const int wave = tid >> 5;          // 0..7
    const int mw   = wave >> 1;         // 0..3  (M subtile row)
    const int nb   = (wave & 1) * 64;   // N base (this wave covers 4 subtiles)
    const int hi   = (lane >> 4) & 1;
    const int l15  = lane & 15;

    __shared__ _Float16 As[64 * 36];   // [m][k]; stride 36 halves (72B, 8B-aligned rows)
    __shared__ _Float16 Bs[128 * 34];  // [n][k] im2col tile (transposed stage)

    v8f acc[4] = {};

    for (int kc = 0; kc < Kpad; kc += 32) {
        // ---- stage A: 64 rows x 32 k of packed weights (64B per row, contiguous)
#if HAVE_ASYNC_LDS
        {
            int q = tid;
            #pragma unroll
            for (int it = 0; it < 2; ++it, q += 256) {   // 512 x 8B chunks
                int r = q >> 3, c8 = q & 7;
                const char* g = (const char*)(wpack + (size_t)(m0 + r) * Kpad + kc) + c8 * 8;
                __builtin_amdgcn_global_load_async_to_lds_b64(
                    (__attribute__((address_space(1))) a_v2i*)g,
                    (__attribute__((address_space(3))) a_v2i*)&As[r * 36 + c8 * 4],
                    0, 0);
            }
        }
#else
        for (int t = tid; t < 64 * 32; t += 256) {
            int r = t >> 5, c = t & 31;
            As[r * 36 + c] = wpack[(size_t)(m0 + r) * Kpad + (kc + c)];
        }
#endif
        // prefetch next chunk's weight rows into cache
        if (kc + 32 < Kpad)
            __builtin_prefetch(wpack + (size_t)(m0 + (tid & 63)) * Kpad + (kc + 32), 0, 0);

        // ---- stage B (virtual im2col), transposed: Bs[n_local][k_local], 128x32
        for (int t = tid; t < 128 * 32; t += 256) {
            int r = t >> 5, c = t & 31;
            int k = kc + c;
            _Float16 val = (_Float16)0.0f;
            if (k < Kreal) {
                int n  = n0 + r;
                int ci = k / (KH * KW);
                int kr = k % (KH * KW);
                int ky = kr / KW, kx = kr % KW;
                int bb = n / S;
                int sr = n % S;
                int oy = sr / OW, ox = sr % OW;
                int iy, ix; bool ok;
                if (!deconv) {
                    iy = oy * stride - pad + ky;
                    ix = ox * stride - pad + kx;
                    ok = (iy >= 0 && iy < H && ix >= 0 && ix < W);
                } else {
                    int ty = oy + pad - ky;
                    int tx = ox + pad - kx;
                    ok = (ty >= 0 && tx >= 0 && (ty % stride) == 0 && (tx % stride) == 0);
                    iy = ty / stride; ix = tx / stride;
                    ok = ok && (iy < H) && (ix < W);
                }
                if (ok) val = in[(((size_t)bb * Cin + ci) * H + iy) * W + ix];
            }
            Bs[r * 34 + c] = val;
        }
#if HAVE_ASYNC_LDS
        __builtin_amdgcn_s_wait_asynccnt(0);
#endif
        __syncthreads();

        // ---- fragments per CDNA5 wave32 16-bit layouts
        v16h afrag;
        {
            const _Float16* Ar = &As[(mw * 16 + l15) * 36];
            #pragma unroll
            for (int i = 0; i < 8; ++i) {
                // A 16x32: vgpr i<4 -> K=2i (+8 for hi lanes); i>=4 -> K=2i+8 (+8 hi)
                int k = (i < 4 ? 2 * i : 2 * i + 8) + hi * 8;
                afrag[2 * i]     = Ar[k];
                afrag[2 * i + 1] = Ar[k + 1];
            }
        }
        v16h bfrag[4];
        #pragma unroll
        for (int j = 0; j < 4; ++j) {
            const _Float16* Br = &Bs[(nb + j * 16 + l15) * 34];
            #pragma unroll
            for (int i = 0; i < 8; ++i) {
                // B 32x16: vgpr i<4 -> K=2i (+8 for hi lanes); i>=4 -> K=2i+8 (+8 hi)
                int k = (i < 4 ? 2 * i : 2 * i + 8) + hi * 8;
                bfrag[j][2 * i]     = Br[k];
                bfrag[j][2 * i + 1] = Br[k + 1];
            }
        }
        #pragma unroll
        for (int j = 0; j < 4; ++j)
            acc[j] = __builtin_amdgcn_wmma_f32_16x16x32_f16(false, afrag, false, bfrag[j],
                                                            (short)0, acc[j], false, false);
        __syncthreads();
    }

    // ---- store: C/D layout — vgpr v holds row M = v + 8*hi; lane l15 = col
    #pragma unroll
    for (int v = 0; v < 8; ++v) {
        int co = m0 + mw * 16 + v + hi * 8;
        if (co < Cout) {
            float bv = bias[co];
            #pragma unroll
            for (int j = 0; j < 4; ++j) {
                int n  = n0 + nb + j * 16 + l15;
                int bb = n / S, sr = n % S;
                out[((size_t)bb * Cout + co) * S + sr] = acc[j][v] + bv;
            }
        }
    }
}

// ---------------------------------------------------------------------------
// Weight packing: f32 (O,I,KH,KW) or deconv (I,O,KH,KW) -> f16 [Mround][Kpad]
// ---------------------------------------------------------------------------
__global__ void pack_w_k(const float* __restrict__ w, _Float16* __restrict__ wp,
                         int Cout, int Cin, int KH, int KW, int Kpad, int Mround, int deconv)
{
    int idx = blockIdx.x * 256 + threadIdx.x;
    int total = Mround * Kpad;
    if (idx >= total) return;
    int co = idx / Kpad, k = idx % Kpad;
    int Kreal = Cin * KH * KW;
    float v = 0.0f;
    if (co < Cout && k < Kreal) {
        int ci = k / (KH * KW);
        int kr = k % (KH * KW);
        int ky = kr / KW, kx = kr % KW;
        if (!deconv) v = w[((size_t)(co * Cin + ci) * KH + ky) * KW + kx];
        else         v = w[((size_t)(ci * Cout + co) * KH + ky) * KW + kx];
    }
    wp[idx] = (_Float16)v;
}

// ---------------------------------------------------------------------------
// BatchNorm batch statistics (biased variance), one WG per channel.
// ---------------------------------------------------------------------------
__global__ __launch_bounds__(256)
void bn_stats_k(const float* __restrict__ x, float* __restrict__ mean,
                float* __restrict__ rstd, int C, int S)
{
    int c = blockIdx.x;
    int total = BB * S;
    double s = 0.0, s2 = 0.0;
    for (int i = threadIdx.x; i < total; i += 256) {
        int bb = i / S, sp = i % S;
        float v = x[((size_t)bb * C + c) * S + sp];
        s += v; s2 += (double)v * (double)v;
    }
    __shared__ double sh[256], sh2[256];
    sh[threadIdx.x] = s; sh2[threadIdx.x] = s2;
    __syncthreads();
    for (int o = 128; o > 0; o >>= 1) {
        if (threadIdx.x < o) { sh[threadIdx.x] += sh[threadIdx.x + o]; sh2[threadIdx.x] += sh2[threadIdx.x + o]; }
        __syncthreads();
    }
    if (threadIdx.x == 0) {
        double m = sh[0] / total;
        double var = sh2[0] / total - m * m;
        mean[c] = (float)m;
        rstd[c] = rsqrtf((float)var + 1e-5f);
    }
}

// BN apply + ReLU (+ optional residual add AFTER relu), emit f32 and/or f16.
__global__ void bn_apply_k(const float* __restrict__ x, const float* __restrict__ mean,
                           const float* __restrict__ rstd, const float* __restrict__ g,
                           const float* __restrict__ be, const float* __restrict__ res,
                           float* __restrict__ o32, _Float16* __restrict__ o16,
                           int C, int S, int total)
{
    int i = blockIdx.x * 256 + threadIdx.x;
    if (i >= total) return;
    int c = (i / S) % C;
    float y = g[c] * (x[i] - mean[c]) * rstd[c] + be[c];
    y = fmaxf(y, 0.0f);
    if (res) y += res[i];
    if (o32) o32[i] = y;
    if (o16) o16[i] = (_Float16)y;
}

__global__ void cvt16_k(const float* __restrict__ x, _Float16* __restrict__ y, int n)
{
    int i = blockIdx.x * 256 + threadIdx.x;
    if (i < n) y[i] = (_Float16)x[i];
}

__global__ void zero_k(float* p) { if (blockIdx.x == 0 && threadIdx.x == 0) p[0] = 0.0f; }

__global__ void esq_k(const float* __restrict__ emb, float* __restrict__ esq, int K)
{
    int k = blockIdx.x * 256 + threadIdx.x;
    if (k >= K) return;
    float s = 0.0f;
    for (int j = 0; j < 64; ++j) { float v = emb[(size_t)k * 64 + j]; s += v * v; }
    esq[k] = s;
}

// ---------------------------------------------------------------------------
// VQ: per-thread 64-d vector, codebook chunked through LDS. Writes closest
// (int32), qo (f16, forward straight-through == emb[closest]) and sum (q-ef)^2.
// ---------------------------------------------------------------------------
__global__ __launch_bounds__(256)
void vq_k(const float* __restrict__ enc, const float* __restrict__ emb,
          const float* __restrict__ esq, int* __restrict__ closest,
          _Float16* __restrict__ qo16, float* __restrict__ lossacc, int K)
{
    int n = blockIdx.x * 256 + threadIdx.x;
    float v[64];
    const float* e = enc + (size_t)n * 64;
    #pragma unroll
    for (int j = 0; j < 64; ++j) v[j] = e[j];

    __shared__ float Em[64 * 64];
    __shared__ float Es[64];
    float best = 3.4e38f; int bi = 0;
    for (int k0 = 0; k0 < K; k0 += 64) {
        __syncthreads();
        for (int t = threadIdx.x; t < 64 * 64; t += 256) Em[t] = emb[(size_t)k0 * 64 + t];
        if (threadIdx.x < 64) Es[threadIdx.x] = esq[k0 + threadIdx.x];
        __syncthreads();
        if (k0 + 64 < K)
            __builtin_prefetch(emb + (size_t)(k0 + 64) * 64 + (threadIdx.x * 16 & 4095), 0, 0);
        for (int kk = 0; kk < 64; ++kk) {
            const float* er = &Em[kk * 64];
            float dot = 0.0f;
            #pragma unroll
            for (int j = 0; j < 64; ++j) dot += v[j] * er[j];
            float d = Es[kk] - 2.0f * dot;   // ||v||^2 is argmin-invariant
            if (d < best) { best = d; bi = k0 + kk; }
        }
    }
    closest[n] = bi;
    const float* q = emb + (size_t)bi * 64;
    float l = 0.0f;
    #pragma unroll
    for (int j = 0; j < 64; ++j) {
        float qv = q[j];
        float df = qv - v[j];
        l += df * df;
        qo16[(size_t)n * 64 + j] = (_Float16)qv;
    }
    atomicAdd(lossacc, l);
}

__global__ void qfin_k(const float* acc, float* out)
{
    if (blockIdx.x == 0 && threadIdx.x == 0)
        out[0] = 1.25f * acc[0] / 2097152.0f;   // (1 + 0.25) * mean over 32*1024*64
}

__global__ void sig_k(const float* __restrict__ x, float* __restrict__ out, int n)
{
    int i = blockIdx.x * 256 + threadIdx.x;
    if (i < n) out[i] = 1.0f / (1.0f + __expf(-x[i]));
}

// ---------------------------------------------------------------------------
// Host orchestration
// ---------------------------------------------------------------------------
extern "C" void kernel_launch(void* const* d_in, const int* in_sizes, int n_in,
                              void* d_out, int out_size, void* d_ws, size_t ws_size,
                              hipStream_t stream)
{
    const float* x     = (const float*)d_in[0];
    const float* emb   = (const float*)d_in[1];
    const float* e_w1  = (const float*)d_in[2];  const float* e_b1 = (const float*)d_in[3];
    const float* e_g1  = (const float*)d_in[4];  const float* e_be1= (const float*)d_in[5];
    const float* e_w2  = (const float*)d_in[6];  const float* e_b2 = (const float*)d_in[7];
    const float* e_g2  = (const float*)d_in[8];  const float* e_be2= (const float*)d_in[9];
    const float* e_rw1 = (const float*)d_in[10]; const float* e_rb1= (const float*)d_in[11];
    const float* e_rg1 = (const float*)d_in[12]; const float* e_rbe1=(const float*)d_in[13];
    const float* e_rw2 = (const float*)d_in[14]; const float* e_rb2= (const float*)d_in[15];
    const float* e_rg2 = (const float*)d_in[16]; const float* e_rbe2=(const float*)d_in[17];
    const float* e_pw  = (const float*)d_in[18]; const float* e_pb = (const float*)d_in[19];
    const float* d_pw  = (const float*)d_in[20]; const float* d_pb = (const float*)d_in[21];
    const float* d_rw1 = (const float*)d_in[22]; const float* d_rb1= (const float*)d_in[23];
    const float* d_rg1 = (const float*)d_in[24]; const float* d_rbe1=(const float*)d_in[25];
    const float* d_rw2 = (const float*)d_in[26]; const float* d_rb2= (const float*)d_in[27];
    const float* d_rg2 = (const float*)d_in[28]; const float* d_rbe2=(const float*)d_in[29];
    const float* d_tw1 = (const float*)d_in[30]; const float* d_tb1= (const float*)d_in[31];
    const float* d_tg1 = (const float*)d_in[32]; const float* d_tbe1=(const float*)d_in[33];
    const float* d_tw2 = (const float*)d_in[34]; const float* d_tb2= (const float*)d_in[35];

    const int Nx   = 32 * 3 * 128 * 128;      // 1,572,864
    const int NE1  = 32 * 256 * 64 * 64;      // 33,554,432
    const int NE2  = 32 * 256 * 32 * 32;      // 8,388,608
    const int Nenc = 32 * 64 * 32 * 32;       // 2,097,152
    const int Nout = Nx;

    // d_out layout: image f32 | closest int32 | qloss f32
    float* out_img    = (float*)d_out;
    int*   out_close  = (int*)((float*)d_out + Nout);
    float* out_qloss  = (float*)d_out + Nout + 32768;

    // workspace bump allocator
    size_t off = 0;
    char* base = (char*)d_ws;
    auto A = [&](size_t bytes) -> void* {
        void* p = base + off;
        off = (off + bytes + 255) & ~(size_t)255;
        return p;
    };
    float* lossacc = (float*)A(4);
    float* esqb    = (float*)A(512 * 4);
    float* bnm     = (float*)A(256 * 4);
    float* bnr     = (float*)A(256 * 4);
    _Float16* x16   = (_Float16*)A((size_t)Nx * 2);
    _Float16* h1_16 = (_Float16*)A((size_t)NE1 * 2);
    float* tmpA  = (float*)A((size_t)NE1 * 4);   // raw conv out at 64x64 (reused)
    float* tmpC  = (float*)A((size_t)NE2 * 4);   // raw conv out at 32x32 (reused)
    float* h2_32 = (float*)A((size_t)NE2 * 4);  _Float16* h2_16 = (_Float16*)A((size_t)NE2 * 2);
    float* h3_32 = (float*)A((size_t)NE2 * 4);  _Float16* h3_16 = (_Float16*)A((size_t)NE2 * 2);
    _Float16* h4_16 = (_Float16*)A((size_t)NE2 * 2);
    float* enc32 = (float*)A((size_t)Nenc * 4);
    _Float16* qo16  = (_Float16*)A((size_t)Nenc * 2);
    float* d0_32 = (float*)A((size_t)NE2 * 4);  _Float16* d0_16 = (_Float16*)A((size_t)NE2 * 2);
    float* d1_32 = (float*)A((size_t)NE2 * 4);  _Float16* d1_16 = (_Float16*)A((size_t)NE2 * 2);
    _Float16* d2_16 = (_Float16*)A((size_t)NE2 * 2);
    _Float16* d3_16 = (_Float16*)A((size_t)NE1 * 2);
    float* outraw = (float*)A((size_t)Nout * 4);

    auto packBytes = [&](int Cout, int Cin, int KH) -> size_t {
        int Kreal = Cin * KH * KH;
        int Kpad = (Kreal + 31) & ~31;
        int Mr = ((Cout + 63) / 64) * 64;
        return (size_t)Mr * Kpad * 2;
    };
    _Float16* pe1  = (_Float16*)A(packBytes(256, 3, 4));
    _Float16* pe2  = (_Float16*)A(packBytes(256, 256, 4));
    _Float16* per1 = (_Float16*)A(packBytes(256, 256, 3));
    _Float16* per2 = (_Float16*)A(packBytes(256, 256, 1));
    _Float16* pep  = (_Float16*)A(packBytes(64, 256, 1));
    _Float16* pdp  = (_Float16*)A(packBytes(256, 64, 1));
    _Float16* pdr1 = (_Float16*)A(packBytes(256, 256, 3));
    _Float16* pdr2 = (_Float16*)A(packBytes(256, 256, 3));
    _Float16* pdt1 = (_Float16*)A(packBytes(256, 256, 4));
    _Float16* pdt2 = (_Float16*)A(packBytes(3, 256, 4));

    auto pack = [&](const float* w, _Float16* wp, int Cout, int Cin, int KH, int deconv) {
        int Kreal = Cin * KH * KH;
        int Kpad = (Kreal + 31) & ~31;
        int Mr = ((Cout + 63) / 64) * 64;
        int tot = Mr * Kpad;
        pack_w_k<<<(tot + 255) / 256, 256, 0, stream>>>(w, wp, Cout, Cin, KH, KH, Kpad, Mr, deconv);
    };
    auto conv = [&](const _Float16* in, const _Float16* wp, const float* bias, float* out,
                    int Cin, int H, int W, int Cout, int OH, int OW,
                    int KH, int stride, int pad, int deconv) {
        int Kreal = Cin * KH * KH;
        int Kpad = (Kreal + 31) & ~31;
        dim3 g((unsigned)(32 * OH * OW / 128), (unsigned)((Cout + 63) / 64));
        conv_wmma_k<<<g, 256, 0, stream>>>(in, wp, bias, out, Cin, H, W, Cout, OH, OW,
                                           KH, KH, stride, pad, Kreal, Kpad, deconv);
    };
    auto bn = [&](const float* raw, const float* g_, const float* be_, const float* res,
                  float* o32, _Float16* o16, int C, int S) {
        bn_stats_k<<<C, 256, 0, stream>>>(raw, bnm, bnr, C, S);
        int tot = 32 * C * S;
        bn_apply_k<<<(tot + 255) / 256, 256, 0, stream>>>(raw, bnm, bnr, g_, be_, res, o32, o16, C, S, tot);
    };

    // ---- prologue ----
    zero_k<<<1, 64, 0, stream>>>(lossacc);
    cvt16_k<<<(Nx + 255) / 256, 256, 0, stream>>>(x, x16, Nx);
    esq_k<<<2, 256, 0, stream>>>(emb, esqb, 512);

    // ---- encoder ----
    pack(e_w1, pe1, 256, 3, 4, 0);
    conv(x16, pe1, e_b1, tmpA, 3, 128, 128, 256, 64, 64, 4, 2, 1, 0);
    bn(tmpA, e_g1, e_be1, nullptr, nullptr, h1_16, 256, 4096);

    pack(e_w2, pe2, 256, 256, 4, 0);
    conv(h1_16, pe2, e_b2, tmpC, 256, 64, 64, 256, 32, 32, 4, 2, 1, 0);
    bn(tmpC, e_g2, e_be2, nullptr, h2_32, h2_16, 256, 1024);

    pack(e_rw1, per1, 256, 256, 3, 0);
    conv(h2_16, per1, e_rb1, tmpC, 256, 32, 32, 256, 32, 32, 3, 1, 1, 0);
    bn(tmpC, e_rg1, e_rbe1, h2_32, h3_32, h3_16, 256, 1024);

    pack(e_rw2, per2, 256, 256, 1, 0);
    conv(h3_16, per2, e_rb2, tmpC, 256, 32, 32, 256, 32, 32, 1, 1, 0, 0);
    bn(tmpC, e_rg2, e_rbe2, h3_32, nullptr, h4_16, 256, 1024);

    pack(e_pw, pep, 64, 256, 1, 0);
    conv(h4_16, pep, e_pb, enc32, 256, 32, 32, 64, 32, 32, 1, 1, 0, 0);

    // ---- VQ ----
    vq_k<<<128, 256, 0, stream>>>(enc32, emb, esqb, out_close, qo16, lossacc, 512);
    qfin_k<<<1, 64, 0, stream>>>(lossacc, out_qloss);

    // ---- decoder ----
    pack(d_pw, pdp, 256, 64, 1, 0);
    conv(qo16, pdp, d_pb, d0_32, 64, 32, 32, 256, 32, 32, 1, 1, 0, 0);
    cvt16_k<<<(NE2 + 255) / 256, 256, 0, stream>>>(d0_32, d0_16, NE2);

    pack(d_rw1, pdr1, 256, 256, 3, 0);
    conv(d0_16, pdr1, d_rb1, tmpC, 256, 32, 32, 256, 32, 32, 3, 1, 1, 0);
    bn(tmpC, d_rg1, d_rbe1, d0_32, d1_32, d1_16, 256, 1024);

    pack(d_rw2, pdr2, 256, 256, 3, 0);
    conv(d1_16, pdr2, d_rb2, tmpC, 256, 32, 32, 256, 32, 32, 3, 1, 1, 0);
    bn(tmpC, d_rg2, d_rbe2, d1_32, nullptr, d2_16, 256, 1024);

    pack(d_tw1, pdt1, 256, 256, 4, 1);
    conv(d2_16, pdt1, d_tb1, tmpA, 256, 32, 32, 256, 64, 64, 4, 2, 1, 1);
    bn(tmpA, d_tg1, d_tbe1, nullptr, nullptr, d3_16, 256, 4096);

    pack(d_tw2, pdt2, 3, 256, 4, 1);
    conv(d3_16, pdt2, d_tb2, outraw, 256, 64, 64, 3, 128, 128, 4, 2, 1, 1);
    sig_k<<<(Nout + 255) / 256, 256, 0, stream>>>(outraw, out_img, Nout);
}
